// ParallelModernBertAttention_20014547599800
// MI455X (gfx1250) — matrix operand order repaired and checked by reading, not compiled, not run on hardware
//
#include <hip/hip_runtime.h>
#include <hip/hip_bf16.h>
#include <math.h>

#define BATCH   2
#define S_LEN   4096
#define HDIM    768
#define NHEADS  12
#define HEADD   64
#define QKV_N   2304
#define HALFWIN 64

typedef __attribute__((ext_vector_type(16))) __bf16 v16bf;
typedef __attribute__((ext_vector_type(2)))  __bf16 bf16x2;
typedef __attribute__((ext_vector_type(8)))  float  v8f;

union BFrag {
    v16bf v;
    unsigned int w[8];
};

// f32 pair -> packed bf16 (lo in [15:0]), RNE. Prefer hw v_cvt_pk_bf16_f32.
__device__ __forceinline__ unsigned int cvt2(float lo, float hi) {
#if __has_builtin(__builtin_amdgcn_cvt_pk_bf16_f32)
    union { bf16x2 v; unsigned int u; } c;
    c.v = __builtin_amdgcn_cvt_pk_bf16_f32(lo, hi);
    return c.u;
#else
    union { float f; unsigned int u; } a, b;
    a.f = lo; b.f = hi;
    unsigned int ua = a.u + 0x7FFFu + ((a.u >> 16) & 1u);
    unsigned int ub = b.u + 0x7FFFu + ((b.u >> 16) & 1u);
    return (ua >> 16) | (ub & 0xFFFF0000u);
#endif
}

__device__ __forceinline__ unsigned short cvt1(float x) {
    return (unsigned short)(cvt2(x, 0.0f) & 0xFFFFu);
}

__device__ __forceinline__ void frag_from(BFrag& f, const uint4& lo, const uint4& hi) {
    f.w[0]=lo.x; f.w[1]=lo.y; f.w[2]=lo.z; f.w[3]=lo.w;
    f.w[4]=hi.x; f.w[5]=hi.y; f.w[6]=hi.z; f.w[7]=hi.w;
}

// ---------------------------------------------------------------------------
// Bulk f32 -> bf16 conversion, 8 elements / thread.
// ---------------------------------------------------------------------------
__global__ void __launch_bounds__(256)
cvt_f32_bf16(const float* __restrict__ src, unsigned short* __restrict__ dst, int n8) {
    int i = blockIdx.x * blockDim.x + threadIdx.x;
    if (i >= n8) return;
    const float4 x0 = ((const float4*)src)[2*i];
    const float4 x1 = ((const float4*)src)[2*i + 1];
    uint4 o;
    o.x = cvt2(x0.x, x0.y); o.y = cvt2(x0.z, x0.w);
    o.z = cvt2(x1.x, x1.y); o.w = cvt2(x1.z, x1.w);
    ((uint4*)dst)[i] = o;
}

// ---------------------------------------------------------------------------
// NT GEMM on pre-converted bf16: C[m,n] = sum_k A[m,k]*B[n,k], f32 accum.
// Software-pipelined with two fully-named register buffers (no arrays ->
// no scratch spills). K must be a multiple of 64.
// One wave computes 16(M) x 64(N). grid=(N/64, M/16), block=32.
// ---------------------------------------------------------------------------
struct RawFrags {
    uint4 a0, a1;
    uint4 b00, b01, b10, b11, b20, b21, b30, b31;
};

__device__ __forceinline__ void load_raw(RawFrags& f, const unsigned int* arow,
                                         const unsigned int* b0, const unsigned int* b1,
                                         const unsigned int* b2, const unsigned int* b3,
                                         int kw, int hf) {
    f.a0  = *(const uint4*)(arow + kw + 8*hf);
    f.a1  = *(const uint4*)(arow + kw + 8*hf + 4);
    f.b00 = *(const uint4*)(b0 + kw + 8*hf);
    f.b01 = *(const uint4*)(b0 + kw + 8*hf + 4);
    f.b10 = *(const uint4*)(b1 + kw + 8*hf);
    f.b11 = *(const uint4*)(b1 + kw + 8*hf + 4);
    f.b20 = *(const uint4*)(b2 + kw + 8*hf);
    f.b21 = *(const uint4*)(b2 + kw + 8*hf + 4);
    f.b30 = *(const uint4*)(b3 + kw + 8*hf);
    f.b31 = *(const uint4*)(b3 + kw + 8*hf + 4);
}

__device__ __forceinline__ void wmma4(const RawFrags& f, v8f& c0, v8f& c1, v8f& c2, v8f& c3) {
    BFrag a, b;
    frag_from(a, f.a0, f.a1);
    frag_from(b, f.b00, f.b01);
    c0 = __builtin_amdgcn_wmma_f32_16x16x32_bf16(false, a.v, false, b.v, (short)0, c0, false, false);
    frag_from(b, f.b10, f.b11);
    c1 = __builtin_amdgcn_wmma_f32_16x16x32_bf16(false, a.v, false, b.v, (short)0, c1, false, false);
    frag_from(b, f.b20, f.b21);
    c2 = __builtin_amdgcn_wmma_f32_16x16x32_bf16(false, a.v, false, b.v, (short)0, c2, false, false);
    frag_from(b, f.b30, f.b31);
    c3 = __builtin_amdgcn_wmma_f32_16x16x32_bf16(false, a.v, false, b.v, (short)0, c3, false, false);
}

__global__ void __launch_bounds__(32)
gemm_nt_bf16(const unsigned short* __restrict__ A, const unsigned short* __restrict__ Bw,
             float* __restrict__ C, int K, int ldc) {
    const int lane = threadIdx.x & 31;
    const int hf   = lane >> 4;
    const int l    = lane & 15;
    const int m0   = blockIdx.y * 16;
    const int n0   = blockIdx.x * 64;

    v8f vzero = {0.f,0.f,0.f,0.f,0.f,0.f,0.f,0.f};
    v8f c0 = vzero, c1 = vzero, c2 = vzero, c3 = vzero;

    const unsigned int* arow = (const unsigned int*)(A + (size_t)(m0 + l) * K);
    const unsigned int* b0 = (const unsigned int*)(Bw + (size_t)(n0 +  0 + l) * K);
    const unsigned int* b1 = (const unsigned int*)(Bw + (size_t)(n0 + 16 + l) * K);
    const unsigned int* b2 = (const unsigned int*)(Bw + (size_t)(n0 + 32 + l) * K);
    const unsigned int* b3 = (const unsigned int*)(Bw + (size_t)(n0 + 48 + l) * K);

    RawFrags f0, f1;
    load_raw(f0, arow, b0, b1, b2, b3, 0, hf);

    #pragma unroll 1
    for (int k0 = 0; k0 < K; k0 += 64) {
        load_raw(f1, arow, b0, b1, b2, b3, (k0 + 32) >> 1, hf);
        wmma4(f0, c0, c1, c2, c3);
        if (k0 + 64 < K)
            load_raw(f0, arow, b0, b1, b2, b3, (k0 + 64) >> 1, hf);
        wmma4(f1, c0, c1, c2, c3);
    }

    #pragma unroll
    for (int r = 0; r < 8; ++r) {
        float* crow = C + (size_t)(m0 + r + 8*hf) * ldc + n0 + l;
        crow[ 0] = c0[r];
        crow[16] = c1[r];
        crow[32] = c2[r];
        crow[48] = c3[r];
    }
}

// ---------------------------------------------------------------------------
// RoPE on q,k (f32 in) -> bf16 qk buffer [B,S,1536]
// ---------------------------------------------------------------------------
__global__ void __launch_bounds__(256)
rope_bf16(const float* __restrict__ qkv, const int* __restrict__ pos_ids,
          unsigned short* __restrict__ qkbf) {
    int idx = blockIdx.x * blockDim.x + threadIdx.x;  // B*S*NH*32
    if (idx >= BATCH * S_LEN * NHEADS * 32) return;
    int d = idx & 31;
    int h = (idx >> 5) % NHEADS;
    int s = (idx / (32 * NHEADS)) % S_LEN;
    int b = idx / (32 * NHEADS * S_LEN);

    float pos = (float)pos_ids[b * S_LEN + s];
    float inv = __powf(10000.0f, -(float)(2 * d) / 64.0f);
    float sn, cs;
    __sincosf(pos * inv, &sn, &cs);

    size_t ibase = (size_t)(b * S_LEN + s) * QKV_N + h * HEADD;
    size_t obase = (size_t)(b * S_LEN + s) * 1536 + h * HEADD;

    float q1 = qkv[ibase + d], q2 = qkv[ibase + d + 32];
    qkbf[obase + d]      = cvt1(q1 * cs - q2 * sn);
    qkbf[obase + d + 32] = cvt1(q2 * cs + q1 * sn);
    float k1 = qkv[ibase + 768 + d], k2 = qkv[ibase + 768 + d + 32];
    qkbf[obase + 768 + d]      = cvt1(k1 * cs - k2 * sn);
    qkbf[obase + 768 + d + 32] = cvt1(k2 * cs + k1 * sn);
}

// ---------------------------------------------------------------------------
// Transpose V: qkv f32 [B,S,2304] (v part) -> vT bf16 [B*NH, 64, S]
// grid=(S/32, 2, B*NH), block=(32,8)
// ---------------------------------------------------------------------------
__global__ void __launch_bounds__(256)
v_transpose(const float* __restrict__ qkv, unsigned short* __restrict__ vT) {
    __shared__ float tile[32][33];
    const int tx = threadIdx.x;
    const int ty = threadIdx.y;
    const int s0 = blockIdx.x * 32;
    const int d0 = blockIdx.y * 32;
    const int bh = blockIdx.z;
    const int b = bh / NHEADS, h = bh % NHEADS;

    const float* vsrc = qkv + (size_t)b * S_LEN * QKV_N + 1536 + h * HEADD;
    #pragma unroll
    for (int j = 0; j < 4; ++j) {
        const int s = s0 + ty + j*8;
        tile[ty + j*8][tx] = vsrc[(size_t)s * QKV_N + d0 + tx];
    }
    __syncthreads();
    unsigned short* vdst = vT + (size_t)bh * HEADD * S_LEN;
    #pragma unroll
    for (int j = 0; j < 4; ++j) {
        const int d = d0 + ty + j*8;
        vdst[(size_t)d * S_LEN + s0 + tx] = cvt1(tile[tx][ty + j*8]);
    }
}

// ---------------------------------------------------------------------------
// Banded flash attention on bf16 q/k/vT, f32 accum, bf16 out [B*S,768].
// grid=(S/16, NH, B), block=32.
// ---------------------------------------------------------------------------
__global__ void __launch_bounds__(32)
attn_kernel(const unsigned short* __restrict__ qkbf,
            const unsigned short* __restrict__ vT,
            unsigned short* __restrict__ attn) {
    __shared__ unsigned short pLds[16 * 32];

    const int lane = threadIdx.x & 31;
    const int hf   = lane >> 4;
    const int l    = lane & 15;
    const int qt   = blockIdx.x;
    const int h    = blockIdx.y;
    const int b    = blockIdx.z;
    const int q0   = qt * 16;

    const unsigned short* qbase = qkbf + (size_t)b * S_LEN * 1536 + h * HEADD;

    // Q fragments (2 K-chunks of HD=64)
    BFrag aQ0, aQ1;
    {
        const unsigned int* qrow =
            (const unsigned int*)(qbase + (size_t)(q0 + l) * 1536);
        uint4 t0 = *(const uint4*)(qrow + 8*hf);
        uint4 t1 = *(const uint4*)(qrow + 8*hf + 4);
        frag_from(aQ0, t0, t1);
        t0 = *(const uint4*)(qrow + 16 + 8*hf);
        t1 = *(const uint4*)(qrow + 16 + 8*hf + 4);
        frag_from(aQ1, t0, t1);
    }

    v8f vzero = {0.f,0.f,0.f,0.f,0.f,0.f,0.f,0.f};
    v8f o0 = vzero, o1 = vzero, o2 = vzero, o3 = vzero;
    float mrow[8], lrow[8];
    #pragma unroll
    for (int r = 0; r < 8; ++r) { mrow[r] = -1e30f; lrow[r] = 0.f; }

    const float scale = 0.125f;  // 1/sqrt(64)
    const size_t vhead = (size_t)(b * NHEADS + h) * HEADD * S_LEN;

    #pragma unroll 1
    for (int pair = 0; pair < 5; ++pair) {
        const int t0 = qt - 4 + 2 * pair;

        // ---- QK^T scores for the two 16-key tiles
        v8f s01[2];
        #pragma unroll
        for (int tt = 0; tt < 2; ++tt) {
            const int kb0  = (t0 + tt) * 16;
            const int krow = kb0 + l;
            const int krc  = krow < 0 ? 0 : (krow >= S_LEN ? S_LEN - 1 : krow);
            const unsigned int* krp =
                (const unsigned int*)(qbase + 768 + (size_t)krc * 1536);
            BFrag bK;
            uint4 u0 = *(const uint4*)(krp + 8*hf);
            uint4 u1 = *(const uint4*)(krp + 8*hf + 4);
            frag_from(bK, u0, u1);
            v8f sc = __builtin_amdgcn_wmma_f32_16x16x32_bf16(
                false, aQ0.v, false, bK.v, (short)0, vzero, false, false);
            u0 = *(const uint4*)(krp + 16 + 8*hf);
            u1 = *(const uint4*)(krp + 16 + 8*hf + 4);
            frag_from(bK, u0, u1);
            sc = __builtin_amdgcn_wmma_f32_16x16x32_bf16(
                false, aQ1.v, false, bK.v, (short)0, sc, false, false);

            v8f sm;
            #pragma unroll
            for (int r = 0; r < 8; ++r) {
                const int qm = q0 + r + 8*hf;
                const int kn = kb0 + l;
                int dist = qm - kn; if (dist < 0) dist = -dist;
                const bool ok = (kn >= 0) && (kn < S_LEN) && (dist <= HALFWIN);
                sm[r] = ok ? sc[r] * scale : -1.0e9f;
            }
            s01[tt] = sm;
        }

        // ---- issue V-fragment loads early (independent of softmax below)
        int kbase = t0 * 16 + 16*hf;
        kbase = kbase < 0 ? 0 : (kbase > S_LEN - 16 ? S_LEN - 16 : kbase);
        const unsigned int* vr0 = (const unsigned int*)(vT + vhead + (size_t)( 0 + l) * S_LEN) + (kbase >> 1);
        const unsigned int* vr1 = (const unsigned int*)(vT + vhead + (size_t)(16 + l) * S_LEN) + (kbase >> 1);
        const unsigned int* vr2 = (const unsigned int*)(vT + vhead + (size_t)(32 + l) * S_LEN) + (kbase >> 1);
        const unsigned int* vr3 = (const unsigned int*)(vT + vhead + (size_t)(48 + l) * S_LEN) + (kbase >> 1);
        const uint4 w0a = ((const uint4*)vr0)[0], w0b = ((const uint4*)vr0)[1];
        const uint4 w1a = ((const uint4*)vr1)[0], w1b = ((const uint4*)vr1)[1];
        const uint4 w2a = ((const uint4*)vr2)[0], w2b = ((const uint4*)vr2)[1];
        const uint4 w3a = ((const uint4*)vr3)[0], w3b = ((const uint4*)vr3)[1];

        // ---- online softmax (row reductions within 16-lane halves)
        float p0[8], p1[8], alpha[8];
        #pragma unroll
        for (int r = 0; r < 8; ++r) {
            float mx = fmaxf(s01[0][r], s01[1][r]);
            mx = fmaxf(mx, __shfl_xor(mx, 1));
            mx = fmaxf(mx, __shfl_xor(mx, 2));
            mx = fmaxf(mx, __shfl_xor(mx, 4));
            mx = fmaxf(mx, __shfl_xor(mx, 8));
            const float nm = fmaxf(mrow[r], mx);
            const float al = __expf(mrow[r] - nm);
            p0[r] = __expf(s01[0][r] - nm);
            p1[r] = __expf(s01[1][r] - nm);
            float rs = p0[r] + p1[r];
            rs += __shfl_xor(rs, 1);
            rs += __shfl_xor(rs, 2);
            rs += __shfl_xor(rs, 4);
            rs += __shfl_xor(rs, 8);
            lrow[r] = lrow[r] * al + rs;
            mrow[r] = nm;
            alpha[r] = al;
        }
        #pragma unroll
        for (int r = 0; r < 8; ++r) {
            o0[r] *= alpha[r]; o1[r] *= alpha[r];
            o2[r] *= alpha[r]; o3[r] *= alpha[r];
        }

        // ---- stage P (bf16) to LDS: row = M, col = key offset 0..31
        #pragma unroll
        for (int r = 0; r < 8; ++r) {
            const int row = r + 8*hf;
            pLds[row * 32 + l]      = cvt1(p0[r]);
            pLds[row * 32 + 16 + l] = cvt1(p1[r]);
        }
        __syncthreads();

        // rebuild P as 16x32 A-fragment
        BFrag aP;
        {
            const unsigned int* pw = (const unsigned int*)pLds;
            const int base = l * 16 + 8*hf;
            aP.w[0] = pw[base + 0];  aP.w[1] = pw[base + 1];
            aP.w[2] = pw[base + 2];  aP.w[3] = pw[base + 3];
            aP.w[4] = pw[base + 4];  aP.w[5] = pw[base + 5];
            aP.w[6] = pw[base + 6]; aP.w[7] = pw[base + 7];
        }

        // ---- PV wmma
        BFrag bV;
        frag_from(bV, w0a, w0b);
        o0 = __builtin_amdgcn_wmma_f32_16x16x32_bf16(false, aP.v, false, bV.v, (short)0, o0, false, false);
        frag_from(bV, w1a, w1b);
        o1 = __builtin_amdgcn_wmma_f32_16x16x32_bf16(false, aP.v, false, bV.v, (short)0, o1, false, false);
        frag_from(bV, w2a, w2b);
        o2 = __builtin_amdgcn_wmma_f32_16x16x32_bf16(false, aP.v, false, bV.v, (short)0, o2, false, false);
        frag_from(bV, w3a, w3b);
        o3 = __builtin_amdgcn_wmma_f32_16x16x32_bf16(false, aP.v, false, bV.v, (short)0, o3, false, false);
        __syncthreads();
    }

    // normalize, write bf16 [B*S, 768]
    #pragma unroll
    for (int r = 0; r < 8; ++r) {
        const int qrow = q0 + r + 8*hf;
        unsigned short* orow = attn + (size_t)(b * S_LEN + qrow) * HDIM + h * HEADD + l;
        const float inv = 1.0f / lrow[r];
        orow[ 0] = cvt1(o0[r] * inv);
        orow[16] = cvt1(o1[r] * inv);
        orow[32] = cvt1(o2[r] * inv);
        orow[48] = cvt1(o3[r] * inv);
    }
}

// ---------------------------------------------------------------------------
extern "C" void kernel_launch(void* const* d_in, const int* in_sizes, int n_in,
                              void* d_out, int out_size, void* d_ws, size_t ws_size,
                              hipStream_t stream) {
    const float* hidden = (const float*)d_in[0];
    const int*   pos    = (const int*)  d_in[3];
    const float* Wqkv   = (const float*)d_in[4];
    const float* Wo     = (const float*)d_in[5];
    float* out = (float*)d_out;

    const size_t nQKV   = (size_t)BATCH * S_LEN * QKV_N;     // f32 qkv
    const size_t nHID   = (size_t)BATCH * S_LEN * HDIM;
    const size_t nWQKV  = (size_t)QKV_N * HDIM;
    const size_t nWO    = (size_t)HDIM * HDIM;
    const size_t nQK    = (size_t)BATCH * S_LEN * 1536;
    const size_t nVT    = (size_t)BATCH * NHEADS * HEADD * S_LEN;

    float* qkv = (float*)d_ws;
    unsigned short* hbf    = (unsigned short*)(qkv + nQKV);
    unsigned short* wqkvbf = hbf + nHID;
    unsigned short* wobf   = wqkvbf + nWQKV;
    unsigned short* qkbf   = wobf + nWO;
    unsigned short* vTbf   = qkbf + nQK;
    unsigned short* attnbf = vTbf + nVT;

    // 0) one-time f32 -> bf16 conversions
    cvt_f32_bf16<<<(int)((nHID/8 + 255)/256), 256, 0, stream>>>(hidden, hbf, (int)(nHID/8));
    cvt_f32_bf16<<<(int)((nWQKV/8 + 255)/256), 256, 0, stream>>>(Wqkv, wqkvbf, (int)(nWQKV/8));
    cvt_f32_bf16<<<(int)((nWO/8 + 255)/256), 256, 0, stream>>>(Wo, wobf, (int)(nWO/8));

    // 1) QKV projection (bf16 x bf16 -> f32)
    {
        dim3 grid(QKV_N / 64, BATCH * S_LEN / 16);
        gemm_nt_bf16<<<grid, 32, 0, stream>>>(hbf, wqkvbf, qkv, HDIM, QKV_N);
    }
    // 2) RoPE -> bf16 q/k ; transpose V -> bf16 vT
    {
        const int n = BATCH * S_LEN * NHEADS * 32;
        rope_bf16<<<(n + 255) / 256, 256, 0, stream>>>(qkv, pos, qkbf);
        dim3 grid(S_LEN / 32, HEADD / 32, BATCH * NHEADS);
        v_transpose<<<grid, dim3(32, 8), 0, stream>>>(qkv, vTbf);
    }
    // 3) Banded flash attention -> bf16 attn
    {
        dim3 grid(S_LEN / 16, NHEADS, BATCH);
        attn_kernel<<<grid, 32, 0, stream>>>(qkbf, vTbf, attnbf);
    }
    // 4) Output projection -> f32 out
    {
        dim3 grid(HDIM / 64, BATCH * S_LEN / 16);
        gemm_nt_bf16<<<grid, 32, 0, stream>>>(attnbf, wobf, out, HDIM, HDIM);
    }
}
